// LiorMemoryState_51410758533299
// MI455X (gfx1250) — compile-verified, matrix operand order checked by reading, not run on hardware
//
#include <hip/hip_runtime.h>
#include <hip/hip_bf16.h>
#include <math.h>

typedef __attribute__((ext_vector_type(16))) _Float16 v16h;
typedef __attribute__((ext_vector_type(8)))  _Float16 v8h;
typedef __attribute__((ext_vector_type(2)))  _Float16 v2h;
typedef __attribute__((ext_vector_type(8)))  float    v8f;
typedef __attribute__((ext_vector_type(4)))  float    f4;

#define B_   4
#define N_   4096
#define D_   1024
#define PEFF 4
#define BM   128
#define BN   256
#define BK   32
#define BKP  40        // padded LDS row stride in halfs (80-byte rows, 16B aligned)
#define KRS  160       // k-window halfs per buffer (320 B, 16B aligned)
#define KSLEN 4256     // reversed padded f16 kernel array: N + 160

__device__ __forceinline__ float sigmoidf_(float x) { return 1.0f / (1.0f + __expf(-x)); }
__device__ __forceinline__ float clip20(float x) { return fminf(fmaxf(x, -20.f), 20.f); }

__device__ __forceinline__ float kval_at(int idx, float alpha, float beta, float gamma,
                                         float delta, float xi, float eta, float omega,
                                         float phi, float zeta, float w0, float w1, float w2) {
  const float dt = fmaxf((float)idx + 1.0f, 0.1f);
  const float ke = alpha * __expf(clip20(-beta * dt));
  const float kf = gamma * __expf(clip20(-delta * __logf(dt))) * __expf(clip20(-xi * dt));
  const float ko = eta * __cosf(omega * dt + phi) * __expf(clip20(-zeta * dt));
  return fminf(fmaxf(w0 * ke + w1 * kf + w2 * ko, -100.f), 100.f);
}

// ---------------------------------------------------------------------------
// Kernel 1: normalized kernel values, written as a REVERSED zero-padded f16
// array ks[j] = kclip[N+31-j] so every GEMM k-window is one contiguous,
// 64B-aligned 160-half slice (no bounds checks / converts in the hot loop).
// ---------------------------------------------------------------------------
__global__ void kvals_kernel(const float* __restrict__ kp,
                             const float* __restrict__ wl,
                             _Float16* __restrict__ ks) {
  __shared__ float red[256];
  const int tid = threadIdx.x;
  const float alpha = __expf(kp[0]);
  const float beta  = __expf(kp[1]);
  const float gamma = __expf(kp[2]);
  const float delta = sigmoidf_(kp[3]);
  const float xi    = __expf(kp[4]);
  const float eta   = __expf(kp[5]);
  const float omega = kp[6];
  const float phi   = kp[7];
  const float zeta  = __expf(kp[8]);
  const float mx = fmaxf(fmaxf(wl[0], wl[1]), wl[2]);
  const float e0 = __expf(wl[0] - mx), e1 = __expf(wl[1] - mx), e2 = __expf(wl[2] - mx);
  const float es = e0 + e1 + e2;
  const float w0 = e0 / es, w1 = e1 / es, w2 = e2 / es;

  float lsum = 0.f;
#pragma unroll
  for (int i = 0; i < 16; ++i)
    lsum += kval_at(tid + i * 256, alpha, beta, gamma, delta, xi, eta, omega, phi, zeta, w0, w1, w2);
  red[tid] = lsum;
  __syncthreads();
  for (int s = 128; s > 0; s >>= 1) {
    if (tid < s) red[tid] += red[tid + s];
    __syncthreads();
  }
  const float inv = 1.0f / (fabsf(red[0]) + 1e-8f);
  for (int j = tid; j < KSLEN; j += 256) {
    const int g = N_ + 31 - j;
    const float v = (g >= 0 && g < N_)
        ? kval_at(g, alpha, beta, gamma, delta, xi, eta, omega, phi, zeta, w0, w1, w2) * inv
        : 0.0f;
    ks[j] = (_Float16)v;
  }
}

// ---------------------------------------------------------------------------
// Stage one K-chunk: 160-half k window (20 x b128) + 32x256 X tile transposed
// into LDS as f16, packing two converted s-rows per ds_store_b32.
// ---------------------------------------------------------------------------
__device__ __forceinline__ void stage_tile(const float* __restrict__ xb,
                                           const _Float16* __restrict__ ks,
                                           _Float16* __restrict__ xsb,
                                           _Float16* __restrict__ krb,
                                           int tid, int s0, int d0, int ksbase) {
  if (tid < 20) {
    const v8h kvv = *(const v8h*)(ks + ksbase + tid * 8);
    *(v8h*)(krb + tid * 8) = kvv;
  }
#pragma unroll
  for (int itr = 0; itr < 4; ++itr) {
    const int l  = tid + itr * 256;
    const int s2 = l >> 6;          // 0..15: pair of s rows
    const int d4 = (l & 63) << 2;   // 0..252
    const float* g0 = xb + (size_t)(s0 + 2 * s2) * D_ + d0 + d4;
    const f4 va = *(const f4*)(g0);
    const f4 vb = *(const f4*)(g0 + D_);
#pragma unroll
    for (int c = 0; c < 4; ++c) {
      v2h p;
      p[0] = (_Float16)va[c];
      p[1] = (_Float16)vb[c];
      *(v2h*)(&xsb[(d4 + c) * BKP + 2 * s2]) = p;
    }
  }
}

// ---------------------------------------------------------------------------
// Kernel 2: causal Toeplitz GEMM  out[b,t,d] = sum_{s<=t} k[t-s] * x[b,s,d]
// Block 128(t) x 256(d), BK=32, 8 wave32s each computing 64x64 via a 4x4 grid
// of v_wmma_f32_16x16x32_f16. Double-buffered LDS, one barrier per K-step.
// ---------------------------------------------------------------------------
__global__ void __launch_bounds__(256)
toeplitz_gemm_kernel(const float* __restrict__ x,
                     const _Float16* __restrict__ ks,
                     float* __restrict__ out) {
  __shared__ __align__(16) _Float16 xs[2 * BN * BKP];  // [buf][d_local][s_local]
  __shared__ __align__(16) _Float16 kr[2 * KRS];       // [buf][reversed k window]

  const int tid   = threadIdx.x;
  const int lane  = tid & 31;
  const int wave  = tid >> 5;
  const int waveT = wave >> 2;   // 0..1  -> 64 t-rows
  const int waveD = wave & 3;    // 0..3  -> 64 d-cols
  const int d0 = blockIdx.x * BN;
  const int t0 = blockIdx.y * BM;
  const int b  = blockIdx.z;

  const float* xb = x + (size_t)b * N_ * D_;

  const int m16    = lane & 15;
  const int Klo    = (lane < 16) ? 0 : 8;   // K split per ISA 16-bit A/B layout
  const int rowOff = (lane >> 4) * 8;       // C/D layout: lanes 16-31 hold M+8

  v8f acc[4][4];
#pragma unroll
  for (int i = 0; i < 4; ++i)
#pragma unroll
    for (int j = 0; j < 4; ++j) acc[i][j] = v8f{};

  const int kEnd  = t0 + BM;        // causality: only s < t0+BM contributes
  const int iters = kEnd >> 5;      // >= 4
  const int ksb0  = N_ - 96 - t0;   // ks window base for s0=0 (always >= 32)

  stage_tile(xb, ks, xs, kr, tid, 0, d0, ksb0);
  __syncthreads();

  for (int it = 0; it < iters; ++it) {
    _Float16* xsb = xs + (it & 1) * (BN * BKP);
    _Float16* krb = kr + (it & 1) * KRS;

    if (it + 1 < iters) {
      const int s1 = (it + 1) << 5;
      stage_tile(xb, ks, xs + ((it + 1) & 1) * (BN * BKP), kr + ((it + 1) & 1) * KRS,
                 tid, s1, d0, ksb0 + s1);
    }
    if (it + 2 < iters) {   // global_prefetch for the chunk after next
      const int sp = (it + 2) << 5;
      __builtin_prefetch(xb + (size_t)(sp + ((tid >> 6) << 1)) * D_ + d0 + ((tid & 63) << 2), 0, 3);
    }

    // --- B fragments: two aligned 16B LDS reads per lane per d-tile
    v16h bfrag[4];
#pragma unroll
    for (int dj = 0; dj < 4; ++dj) {
      const int nl = waveD * 64 + dj * 16 + m16;
      const v8h lo = *(const v8h*)(&xsb[nl * BKP + Klo]);
      const v8h hi = *(const v8h*)(&xsb[nl * BKP + Klo + 16]);
      v16h bf;
#pragma unroll
      for (int j = 0; j < 8; ++j) { bf[j] = lo[j]; bf[8 + j] = hi[j]; }
      bfrag[dj] = bf;
    }

    // --- A (Toeplitz) fragments from reversed k window + 16 WMMAs
#pragma unroll
    for (int mi = 0; mi < 4; ++mi) {
      const int tl   = waveT * 64 + mi * 16 + m16;
      const int base = 127 - tl + Klo;
      v16h af;
#pragma unroll
      for (int j = 0; j < 8; ++j) {
        af[j]     = krb[base + j];
        af[8 + j] = krb[base + 16 + j];
      }
#pragma unroll
      for (int dj = 0; dj < 4; ++dj)
        acc[mi][dj] = __builtin_amdgcn_wmma_f32_16x16x32_f16(
            false, af, false, bfrag[dj], (short)0, acc[mi][dj], false, false);
    }
    __syncthreads();
  }

  // --- epilogue: C layout VGPR r -> row M=r (+8 for lanes 16-31), col N=lane%16
#pragma unroll
  for (int mi = 0; mi < 4; ++mi)
#pragma unroll
    for (int dj = 0; dj < 4; ++dj) {
      const int dcol = d0 + waveD * 64 + dj * 16 + m16;
#pragma unroll
      for (int r = 0; r < 8; ++r) {
        const int t = t0 + waveT * 64 + mi * 16 + r + rowOff;
        out[((size_t)b * N_ + t) * D_ + dcol] = acc[mi][dj][r];
      }
    }
}

// ---------------------------------------------------------------------------
// Kernel 3: m recurrence + buffer_new + time-mean of x (fused, x read once)
// ---------------------------------------------------------------------------
__global__ void mfinal_kernel(const float* __restrict__ x,
                              const float* __restrict__ m_in,
                              const float* __restrict__ buffer,
                              const float* __restrict__ j_h,
                              const float* __restrict__ kp,
                              float* __restrict__ m_out,
                              float* __restrict__ buf_out,
                              float* __restrict__ mean_out) {
  const int gid = blockIdx.x * blockDim.x + threadIdx.x;
  if (gid >= B_ * D_) return;
  const int b = gid / D_;
  const int d = gid % D_;
  const float jh = j_h[b];
  const float th = tanhf(jh);
  const float rho   = sigmoidf_(kp[9]) * sigmoidf_(jh);
  const float eta_r = __expf(kp[10]) * (1.0f + 0.1f * th);
  const float xi_r  = __expf(kp[11]) * (1.0f + 0.1f * th);
  const float* xb = x + (size_t)b * N_ * D_ + d;

  float ring[4];
#pragma unroll
  for (int p = 0; p < 4; ++p) ring[p] = buffer[(size_t)(b * PEFF + p) * D_ + d];

  float m = m_in[gid];
  float sum = 0.f;
  for (int t = 0; t < N_; ++t) {
    const float xc = xb[(size_t)t * D_];
    const float xd = ring[t & 3];       // x[t-4] (or initial buffer for t<4)
    m = rho * m + eta_r * xc - xi_r * xd;
    sum += xc;
    ring[t & 3] = xc;
  }
  m_out[gid] = m;
  mean_out[gid] = sum * (1.0f / (float)N_);
#pragma unroll
  for (int p = 0; p < 4; ++p)
    buf_out[(size_t)(b * PEFF + p) * D_ + d] = xb[(size_t)(N_ - PEFF + p) * D_];
}

// ---------------------------------------------------------------------------
// Kernel 4: j_h_new = gelu(mean @ w1 + b1) @ w2 + b2   (tiny: 1 block)
// ---------------------------------------------------------------------------
__global__ void jh_kernel(const float* __restrict__ mean,
                          const float* __restrict__ w1,
                          const float* __restrict__ b1,
                          const float* __restrict__ w2,
                          const float* __restrict__ b2,
                          float* __restrict__ jh_out) {
  __shared__ float red[1024];
  const int tid = threadIdx.x;  // 1024 threads: b = tid/256, j = tid%256
  const int b = tid >> 8;
  const int j = tid & 255;
  const float* mb = mean + b * D_;
  float acc = b1[j];
  for (int i = 0; i < D_; ++i) acc = fmaf(mb[i], w1[i * (D_ / 4) + j], acc);
  const float h = 0.5f * acc * (1.0f + erff(acc * 0.70710678118654752f));  // exact gelu
  red[tid] = h * w2[j];
  __syncthreads();
  for (int s = 128; s > 0; s >>= 1) {
    if (j < s) red[tid] += red[tid + s];
    __syncthreads();
  }
  if (j == 0) jh_out[b] = red[tid] + b2[0];
}

// ---------------------------------------------------------------------------
extern "C" void kernel_launch(void* const* d_in, const int* in_sizes, int n_in,
                              void* d_out, int out_size, void* d_ws, size_t ws_size,
                              hipStream_t stream) {
  const float* x   = (const float*)d_in[0];
  const float* m   = (const float*)d_in[1];
  const float* buf = (const float*)d_in[2];
  const float* jh  = (const float*)d_in[3];
  const float* kp  = (const float*)d_in[4];
  const float* wl  = (const float*)d_in[5];
  const float* w1  = (const float*)d_in[6];
  const float* b1  = (const float*)d_in[7];
  const float* w2  = (const float*)d_in[8];
  const float* b2  = (const float*)d_in[9];

  float* out = (float*)d_out;
  // output tuple layout (flat, return order)
  const size_t OUT_M   = (size_t)B_ * N_ * D_;             // (B,D)
  const size_t OUT_BUF = OUT_M + (size_t)B_ * D_;          // (B,PEFF,D)
  const size_t OUT_JH  = OUT_BUF + (size_t)B_ * PEFF * D_; // (B,1)

  _Float16* ks = (_Float16*)d_ws;                          // KSLEN halfs (8512 B)
  float* mean  = (float*)((char*)d_ws + (size_t)KSLEN * sizeof(_Float16)); // B_*D_ floats

  kvals_kernel<<<1, 256, 0, stream>>>(kp, wl, ks);

  dim3 grid(D_ / BN, N_ / BM, B_);
  toeplitz_gemm_kernel<<<grid, 256, 0, stream>>>(x, ks, out);

  mfinal_kernel<<<(B_ * D_) / 256, 256, 0, stream>>>(x, m, buf, jh, kp,
                                                     out + OUT_M, out + OUT_BUF, mean);

  jh_kernel<<<1, 1024, 0, stream>>>(mean, w1, b1, w2, b2, out + OUT_JH);
}
